// CohereAttention_20280835571778
// MI455X (gfx1250) — matrix operand order repaired and checked by reading, not compiled, not run on hardware
//
#include <hip/hip_runtime.h>
#include <cstdint>
#include <cstddef>

typedef __bf16 bf16_t;
typedef __bf16 v16bf __attribute__((ext_vector_type(16)));
typedef float  v8f   __attribute__((ext_vector_type(8)));

namespace {
constexpr int Bc = 2, Sc = 2048, Hc = 32, KVHc = 8, Dc = 128, HIDc = 4096;
constexpr int WIN = 512;
constexpr int QKVN = Hc * Dc + 2 * KVHc * Dc;     // 6144
constexpr float SCALEc = 0.08838834764831845f;    // 128^-0.5
constexpr float EPSc = 1e-5f;
constexpr int BSTRIDE = 40;                       // padded LDS k-stride for B tiles
}

// Fragment union: v16bf for WMMA operand, float4 pair for 2x16B vector loads.
union Frag {
  v16bf v;
  float4 q[2];
  bf16_t e[16];
};

__device__ __forceinline__ float rmax16(float v) {
#pragma unroll
  for (int m = 1; m < 16; m <<= 1) v = fmaxf(v, __shfl_xor(v, m, 32));
  return v;
}
__device__ __forceinline__ float rsum16(float v) {
#pragma unroll
  for (int m = 1; m < 16; m <<= 1) v += __shfl_xor(v, m, 32);
  return v;
}
__device__ __forceinline__ float rsum32(float v) {
#pragma unroll
  for (int m = 1; m < 32; m <<= 1) v += __shfl_xor(v, m, 32);
  return v;
}

// CDNA5 async global->LDS copy (bypasses VGPRs, tracked with ASYNCcnt).
// GVS mode: global addr = saddr(base) + vaddr(32-bit byte offset) + inst_offset,
// LDS addr = VDST vgpr + inst_offset (inst_offset applies to both sides).
__device__ __forceinline__ void async_g2l_b128(unsigned lds_off, unsigned g_off,
                                               const void* base) {
  asm volatile("global_load_async_to_lds_b128 %0, %1, %2"
               :: "v"(lds_off), "v"(g_off), "s"(base) : "memory");
}
__device__ __forceinline__ void async_g2l_b128_o16(unsigned lds_off, unsigned g_off,
                                                   const void* base) {
  asm volatile("global_load_async_to_lds_b128 %0, %1, %2 offset:16"
               :: "v"(lds_off), "v"(g_off), "s"(base) : "memory");
}
__device__ __forceinline__ void wait_async0() {
  asm volatile("s_wait_asynccnt 0x0" ::: "memory");
}

// ---------------- conversion / packing ----------------
__global__ __launch_bounds__(256) void k_f32_to_bf16(const float* __restrict__ src,
                                                     bf16_t* __restrict__ dst, int n) {
  int i = blockIdx.x * 256 + threadIdx.x;
  const int stride = gridDim.x * 256;
  for (; i < n; i += stride) dst[i] = (bf16_t)src[i];
}

// Pack {wq|wk|wv} transposed: dst[n][k] = W[k][n], n in [0,6144), k in [0,4096).
__global__ __launch_bounds__(256) void k_pack_wqkv_T(const float* __restrict__ wq,
                                                     const float* __restrict__ wk,
                                                     const float* __restrict__ wv,
                                                     bf16_t* __restrict__ dst) {
  int i = blockIdx.x * 256 + threadIdx.x;
  const int total = QKVN * HIDc;
  const int stride = gridDim.x * 256;
  for (; i < total; i += stride) {
    const int n = i / HIDc, k = i % HIDc;
    float v;
    if (n < Hc * Dc)                  v = wq[(size_t)k * (Hc * Dc) + n];
    else if (n < Hc * Dc + KVHc * Dc) v = wk[(size_t)k * (KVHc * Dc) + (n - Hc * Dc)];
    else                              v = wv[(size_t)k * (KVHc * Dc) + (n - Hc * Dc - KVHc * Dc)];
    dst[i] = (bf16_t)v;
  }
}

// Transpose + convert: dst[n][k] = src[k][n]; src is [K][N] f32.
__global__ __launch_bounds__(256) void k_transpose_to_bf16(const float* __restrict__ src,
                                                           bf16_t* __restrict__ dst,
                                                           int K, int N) {
  int i = blockIdx.x * 256 + threadIdx.x;
  const int total = K * N;
  const int stride = gridDim.x * 256;
  for (; i < total; i += stride) {
    const int n = i / K, k = i % K;
    dst[i] = (bf16_t)src[(size_t)k * N + n];
  }
}

// ---------------- bf16 WMMA GEMM: C[M,N] = A[M,K] * B[K,N], B given transposed [N][K] ----------------
// 128x128 tile per 256-thread (8-wave) workgroup; 32x64 per wave (2 A-frags x 4 B-frags = 8 WMMA
// per K-step of 32). Double-buffered LDS, staged with GLOBAL_LOAD_ASYNC_TO_LDS_B128 overlapping
// the WMMA block; one s_wait_asynccnt + barrier per K-step.
__global__ __launch_bounds__(256) void k_gemm_bf16(const bf16_t* __restrict__ A,
                                                   const bf16_t* __restrict__ BT,
                                                   float* __restrict__ Cf,
                                                   bf16_t* __restrict__ Cb,
                                                   int M, int N, int K, int out_bf16) {
  __shared__ bf16_t As[2][128 * 32];        // [row][k] k-contiguous
  __shared__ bf16_t Bt[2][128 * BSTRIDE];   // [col][k] k-contiguous, padded stride

  const int t = threadIdx.x;
  const int lane = t & 31;
  const int w = t >> 5;
  const int l15 = lane & 15;
  const int lh = lane >> 4;
  const int kb8 = lh * 8;

  const int m0 = blockIdx.y * 128;
  const int n0 = blockIdx.x * 128;
  const int wr = (w >> 1) * 32;      // wave's 32-row band
  const int wc = (w & 1) * 64;       // wave's 64-col band

  v8f acc[2][4] = {};

  // staging coords: thread handles one 16-element half-row (two async b128 per matrix)
  const int srow = t >> 1;           // 0..127
  const int skh  = (t & 1) * 16;     // 0 or 16

  // per-lane LDS byte addresses (low 32 bits of a generic pointer into LDS = LDS offset)
  const unsigned aLds[2] = {(unsigned)(uintptr_t)&As[0][srow * 32 + skh],
                            (unsigned)(uintptr_t)&As[1][srow * 32 + skh]};
  const unsigned bLds[2] = {(unsigned)(uintptr_t)&Bt[0][srow * BSTRIDE + skh],
                            (unsigned)(uintptr_t)&Bt[1][srow * BSTRIDE + skh]};
  // per-lane global byte offsets at k=0
  const unsigned aOff0 = (unsigned)(((size_t)(m0 + srow) * K + skh) * sizeof(bf16_t));
  const unsigned bOff0 = (unsigned)(((size_t)(n0 + srow) * K + skh) * sizeof(bf16_t));

  // prologue: stage K-step 0 into buffer 0
  async_g2l_b128(aLds[0], aOff0, A);
  async_g2l_b128_o16(aLds[0], aOff0, A);
  async_g2l_b128(bLds[0], bOff0, BT);
  async_g2l_b128_o16(bLds[0], bOff0, BT);
  wait_async0();
  __syncthreads();

  int buf = 0;
  for (int k0 = 0; k0 < K; k0 += 32, buf ^= 1) {
    // kick off async staging of the next K-step into the other buffer
    if (k0 + 32 < K) {
      const unsigned kadd = (unsigned)((k0 + 32) * sizeof(bf16_t));
      async_g2l_b128(aLds[buf ^ 1], aOff0 + kadd, A);
      async_g2l_b128_o16(aLds[buf ^ 1], aOff0 + kadd, A);
      async_g2l_b128(bLds[buf ^ 1], bOff0 + kadd, BT);
      async_g2l_b128_o16(bLds[buf ^ 1], bOff0 + kadd, BT);
    }

    // A fragments for the wave's two 16-row sub-bands
    Frag a0, a1;
    a0.q[0] = *(const float4*)(&As[buf][(wr + l15) * 32 + kb8]);
    a0.q[1] = *(const float4*)(&As[buf][(wr + l15) * 32 + 16 + kb8]);
    a1.q[0] = *(const float4*)(&As[buf][(wr + 16 + l15) * 32 + kb8]);
    a1.q[1] = *(const float4*)(&As[buf][(wr + 16 + l15) * 32 + 16 + kb8]);

#pragma unroll
    for (int s = 0; s < 4; ++s) {
      Frag bfr;   // B fragment: lane holds col l15 of subtile; K = 8*lh + e, 16 + 8*lh + (e-8) (table)
      const int col = wc + s * 16 + l15;
      bfr.q[0] = *(const float4*)(&Bt[buf][col * BSTRIDE + kb8]);
      bfr.q[1] = *(const float4*)(&Bt[buf][col * BSTRIDE + 16 + kb8]);
      acc[0][s] = __builtin_amdgcn_wmma_f32_16x16x32_bf16(false, a0.v, false, bfr.v,
                                                          (short)0, acc[0][s], false, false);
      acc[1][s] = __builtin_amdgcn_wmma_f32_16x16x32_bf16(false, a1.v, false, bfr.v,
                                                          (short)0, acc[1][s], false, false);
    }

    // drain this step's async copies (they overlapped the 8 WMMAs above), then sync
    wait_async0();
    __syncthreads();
  }

  // C fragment: vgpr r holds row sub*16 + r + 8*lh, col = l15
#pragma unroll
  for (int sub = 0; sub < 2; ++sub) {
#pragma unroll
    for (int s = 0; s < 4; ++s) {
#pragma unroll
      for (int vr = 0; vr < 8; ++vr) {
        const int row = m0 + wr + sub * 16 + vr + lh * 8;
        const int col = n0 + wc + s * 16 + l15;
        const float v = acc[sub][s][vr];
        if (out_bf16) Cb[(size_t)row * N + col] = (bf16_t)v;
        else          Cf[(size_t)row * N + col] = v;
      }
    }
  }
}

// ---------------- fused LayerNorm + interleaved RoPE + head-major scatter ----------------
// Q -> [B][H][S][D], K -> [B][KVH][S][D], V -> transposed [B][KVH][D][S].
__global__ __launch_bounds__(32) void k_lnrope(const bf16_t* __restrict__ qkv,
                                               const float* __restrict__ cosb,
                                               const float* __restrict__ sinb,
                                               const int* __restrict__ pos,
                                               const float* __restrict__ qw,
                                               const float* __restrict__ kw,
                                               bf16_t* __restrict__ qbuf,
                                               bf16_t* __restrict__ kbuf,
                                               bf16_t* __restrict__ vbufT) {
  const int row = blockIdx.x;           // b*S + s
  const int hh  = blockIdx.y;           // 0..H+2*KVH-1
  const int b = row / Sc, s = row % Sc;
  const int t = threadIdx.x;
  const int d0 = t * 4;

  int cbase;
  if (hh < Hc)               cbase = hh * Dc;
  else if (hh < Hc + KVHc)   cbase = Hc * Dc + (hh - Hc) * Dc;
  else                       cbase = Hc * Dc + KVHc * Dc + (hh - Hc - KVHc) * Dc;

  const bf16_t* x = qkv + (size_t)row * QKVN + cbase;
  float v[4];
#pragma unroll
  for (int i = 0; i < 4; ++i) v[i] = (float)x[d0 + i];

  if (hh >= Hc + KVHc) {                // V head: convert + transposed scatter
    const int kv = hh - Hc - KVHc;
    bf16_t* base = vbufT + (size_t)(b * KVHc + kv) * Dc * Sc;
#pragma unroll
    for (int i = 0; i < 4; ++i) base[(size_t)(d0 + i) * Sc + s] = (bf16_t)v[i];
    return;
  }

  float sum = v[0] + v[1] + v[2] + v[3];
  sum = rsum32(sum);
  const float mu = sum * (1.0f / Dc);
  float vs = 0.f;
#pragma unroll
  for (int i = 0; i < 4; ++i) { const float d = v[i] - mu; vs += d * d; }
  vs = rsum32(vs);
  const float inv = rsqrtf(vs * (1.0f / Dc) + EPSc);
  const float* w = (hh < Hc) ? qw : kw;
  float y[4];
#pragma unroll
  for (int i = 0; i < 4; ++i) y[i] = w[d0 + i] * (v[i] - mu) * inv;

  const int p = pos[row];
  float o[4];
#pragma unroll
  for (int j = 0; j < 4; j += 2) {
    const float c0 = cosb[(size_t)p * Dc + d0 + j];
    const float s0 = sinb[(size_t)p * Dc + d0 + j];
    const float c1 = cosb[(size_t)p * Dc + d0 + j + 1];
    const float s1 = sinb[(size_t)p * Dc + d0 + j + 1];
    o[j]     = y[j] * c0 - y[j + 1] * s0;
    o[j + 1] = y[j + 1] * c1 + y[j] * s1;
  }
  bf16_t* dst;
  if (hh < Hc) dst = qbuf + ((size_t)(b * Hc + hh) * Sc + s) * Dc + d0;
  else         dst = kbuf + ((size_t)(b * KVHc + (hh - Hc)) * Sc + s) * Dc + d0;
#pragma unroll
  for (int i = 0; i < 4; ++i) dst[i] = (bf16_t)o[i];
}

// ---------------- flash-style sliding-window attention (1 wave / 16 queries) ----------------
__global__ __launch_bounds__(32) void k_attn(const bf16_t* __restrict__ qbuf,
                                             const bf16_t* __restrict__ kbuf,
                                             const bf16_t* __restrict__ vbufT,
                                             bf16_t* __restrict__ obuf) {
  const int q0 = blockIdx.x * 16;
  const int h  = blockIdx.y;
  const int b  = blockIdx.z;
  const int hk = h >> 2;                         // H/KVH = 4
  const int lane = threadIdx.x;
  const int l15 = lane & 15;
  const int lh = lane >> 4;
  const int kb8 = lh * 8;

  const bf16_t* Qp = qbuf  + (size_t)(b * Hc + h)    * Sc * Dc;
  const bf16_t* Kp = kbuf  + (size_t)(b * KVHc + hk) * Sc * Dc;
  const bf16_t* Vt = vbufT + (size_t)(b * KVHc + hk) * Dc * Sc;   // [D][S]

  __shared__ bf16_t ps[16 * 32];                 // P tile staging (C-frag -> A-frag reshape)

  // Q A-fragments for the 4 K=32 chunks of D=128
  Frag qf[4];
  const int qrow = q0 + l15;
#pragma unroll
  for (int kc = 0; kc < 4; ++kc) {
    qf[kc].q[0] = *(const float4*)(Qp + (size_t)qrow * Dc + kc * 32 + kb8);
    qf[kc].q[1] = *(const float4*)(Qp + (size_t)qrow * Dc + kc * 32 + 16 + kb8);
  }

  v8f oacc[8] = {};
  float mrow[8], lsum[8];
#pragma unroll
  for (int i = 0; i < 8; ++i) { mrow[i] = -3.0e38f; lsum[i] = 0.f; }

  int jlo = q0 - (WIN - 1);
  if (jlo < 0) jlo = 0;
  jlo &= ~31;
  // j0 is 32-aligned and j0 <= q0+15 <= 2047, so j0+31 <= 2047: no clamping needed.

  for (int j0 = jlo; j0 <= q0 + 15; j0 += 32) {
    // ---- scores S[16q x 32k] = Q * K^T ----
    v8f sc[2] = {};
#pragma unroll
    for (int nt = 0; nt < 2; ++nt) {
      const int key = j0 + nt * 16 + l15;
      const bf16_t* kp = Kp + (size_t)key * Dc;
#pragma unroll
      for (int kc = 0; kc < 4; ++kc) {
        Frag kf;   // B-frag: col=key, contraction d = kc*32 + table order (8*lh + e, 16 + 8*lh + e-8)
        kf.q[0] = *(const float4*)(kp + kc * 32 + kb8);
        kf.q[1] = *(const float4*)(kp + kc * 32 + 16 + kb8);
        sc[nt] = __builtin_amdgcn_wmma_f32_16x16x32_bf16(false, qf[kc].v, false, kf.v,
                                                         (short)0, sc[nt], false, false);
      }
    }

    // ---- online softmax over this 32-key step ----
    float pv0[8], pv1[8];
#pragma unroll
    for (int vr = 0; vr < 8; ++vr) {
      const int r = vr + lh * 8;
      const int q = q0 + r;
      const int key0 = j0 + l15;
      const int key1 = j0 + 16 + l15;
      float s0 = sc[0][vr] * SCALEc;
      float s1 = sc[1][vr] * SCALEc;
      const bool ok0 = (key0 <= q) && (q - key0 < WIN);
      const bool ok1 = (key1 <= q) && (q - key1 < WIN);
      s0 = ok0 ? s0 : -3.0e38f;
      s1 = ok1 ? s1 : -3.0e38f;
      const float rm = rmax16(fmaxf(s0, s1));
      const float mn = fmaxf(mrow[vr], rm);
      const float fac = __expf(mrow[vr] - mn);
      mrow[vr] = mn;
      const float p0 = ok0 ? __expf(s0 - mn) : 0.f;
      const float p1 = ok1 ? __expf(s1 - mn) : 0.f;
      lsum[vr] = lsum[vr] * fac + rsum16(p0 + p1);
#pragma unroll
      for (int dt = 0; dt < 8; ++dt) oacc[dt][vr] = oacc[dt][vr] * fac;
      pv0[vr] = p0;
      pv1[vr] = p1;
    }

    // ---- reshape P (C-frag layout) -> A-frag via LDS ----
    __syncthreads();
#pragma unroll
    for (int vr = 0; vr < 8; ++vr) {
      const int r = vr + lh * 8;
      ps[r * 32 + l15]      = (bf16_t)pv0[vr];
      ps[r * 32 + 16 + l15] = (bf16_t)pv1[vr];
    }
    __syncthreads();

    Frag pf;
    pf.q[0] = *(const float4*)(ps + l15 * 32 + kb8);
    pf.q[1] = *(const float4*)(ps + l15 * 32 + 16 + kb8);

    // ---- O += P * V : V read from transposed layout, key-contiguous ----
#pragma unroll
    for (int dt = 0; dt < 8; ++dt) {
      Frag vf;   // B-frag: col = d, contraction key = j0 + table order (8*lh + e, 16 + 8*lh + e-8)
      const int d = dt * 16 + l15;
      const bf16_t* vp = Vt + (size_t)d * Sc + j0 + kb8;
      vf.q[0] = *(const float4*)(vp);
      vf.q[1] = *(const float4*)(vp + 16);
      oacc[dt] = __builtin_amdgcn_wmma_f32_16x16x32_bf16(false, pf.v, false, vf.v,
                                                         (short)0, oacc[dt], false, false);
    }
  }

  // ---- normalize and scatter to [B, S, H*D] bf16 ----
#pragma unroll
  for (int vr = 0; vr < 8; ++vr) {
    const int q = q0 + vr + lh * 8;
    const float inv = 1.0f / lsum[vr];
#pragma unroll
    for (int dt = 0; dt < 8; ++dt) {
      obuf[((size_t)b * Sc + q) * (Hc * Dc) + (size_t)h * Dc + dt * 16 + l15] =
          (bf16_t)(oacc[dt][vr] * inv);
    }
  }
}

// ---------------- host-side orchestration ----------------
extern "C" void kernel_launch(void* const* d_in, const int* in_sizes, int n_in,
                              void* d_out, int out_size, void* d_ws, size_t ws_size,
                              hipStream_t stream) {
  (void)in_sizes; (void)n_in; (void)out_size; (void)ws_size;
  const float* hidden = (const float*)d_in[0];
  const int*   pos    = (const int*)d_in[1];
  const float* cosb   = (const float*)d_in[2];
  const float* sinb   = (const float*)d_in[3];
  const float* wq     = (const float*)d_in[4];
  const float* wk     = (const float*)d_in[5];
  const float* wv     = (const float*)d_in[6];
  const float* wo     = (const float*)d_in[7];
  const float* qnw    = (const float*)d_in[8];
  const float* knw    = (const float*)d_in[9];
  float* out = (float*)d_out;

  char* ws = (char*)d_ws;
  size_t off = 0;
  auto wsalloc = [&](size_t bytes) {
    char* p = ws + off;
    off += (bytes + 255) & ~(size_t)255;
    return p;
  };

  const int M = Bc * Sc;                                  // 4096 tokens
  bf16_t* Xb     = (bf16_t*)wsalloc((size_t)M * HIDc * 2);
  bf16_t* WqkvT  = (bf16_t*)wsalloc((size_t)QKVN * HIDc * 2);   // [6144][4096]
  bf16_t* WoT    = (bf16_t*)wsalloc((size_t)HIDc * HIDc * 2);   // [4096][4096]
  bf16_t* QKVb   = (bf16_t*)wsalloc((size_t)M * QKVN * 2);
  bf16_t* Qb     = (bf16_t*)wsalloc((size_t)Bc * Hc * Sc * Dc * 2);
  bf16_t* Kb     = (bf16_t*)wsalloc((size_t)Bc * KVHc * Sc * Dc * 2);
  bf16_t* VbT    = (bf16_t*)wsalloc((size_t)Bc * KVHc * Dc * Sc * 2);
  bf16_t* Ab     = (bf16_t*)wsalloc((size_t)M * (Hc * Dc) * 2);

  // 1) precision conversion / transposed weight packing
  k_f32_to_bf16<<<4096, 256, 0, stream>>>(hidden, Xb, M * HIDc);
  k_pack_wqkv_T<<<4096, 256, 0, stream>>>(wq, wk, wv, WqkvT);
  k_transpose_to_bf16<<<4096, 256, 0, stream>>>(wo, WoT, HIDc, HIDc);

  // 2) fused QKV projection: [4096,4096] x [4096,6144] -> bf16
  k_gemm_bf16<<<dim3(QKVN / 128, M / 128), 256, 0, stream>>>(
      Xb, WqkvT, nullptr, QKVb, M, QKVN, HIDc, 1);

  // 3) LayerNorm(Q,K) + RoPE + head-major scatter (V: convert + transpose)
  k_lnrope<<<dim3(M, Hc + 2 * KVHc), 32, 0, stream>>>(
      QKVb, cosb, sinb, pos, qnw, knw, Qb, Kb, VbT);

  // 4) sliding-window flash attention
  k_attn<<<dim3(Sc / 16, Hc, Bc), 32, 0, stream>>>(Qb, Kb, VbT, Ab);

  // 5) output projection: [4096,4096] x [4096,4096] -> f32 d_out
  k_gemm_bf16<<<dim3(HIDc / 128, M / 128), 256, 0, stream>>>(
      Ab, WoT, out, nullptr, M, HIDc, HIDc, 0);
}
